// VanillaLSTM_83399674954240
// MI455X (gfx1250) — matrix operand order repaired and checked by reading, not compiled, not run on hardware
//
#include <hip/hip_runtime.h>
#include <math.h>

typedef __attribute__((ext_vector_type(8)))  _Float16 v8h;
typedef __attribute__((ext_vector_type(16))) _Float16 v16h;
typedef __attribute__((ext_vector_type(8)))  float    v8f;
typedef __attribute__((ext_vector_type(2)))  float    f2v;

#define T_  512
#define F_  128
#define H_  512
#define B_  256
#define K_  640          // F_ + H_
#define N_  2048         // 4*H_
#define KT  20           // K_/32 wmma k-steps
#define NT  128          // N_/16 wmma n-tiles
#define LDA 648          // padded A row stride (halves): 648*2B=1296B=16B*81, rows hit distinct banks

union V16 { v16h v; v8h h[2]; };

__device__ __forceinline__ float sigmoidf_(float z) {
  return 1.0f / (1.0f + __expf(-z));
}

// -------------------------------------------------------------------------
// One-shot pre-swizzle: Wc = [W;U] (640x2048 f32, row-major) -> fp16 WMMA
// B-fragments laid out so each lane reads one contiguous 32B block per tile:
//   wq[((kt*NT + nt)*32 + lane)*16 + kk]
//     = Wc[kt*32 + (lane<16 ? 0 : 8) + (kk&7) + (kk&8 ? 16 : 0)][nt*16 + (lane&15)]
// (B 32x16 fp16 layout: lanes 0-15 hold K=0..7,16..23, lanes 16-31 hold K=8..15,24..31,
//  16 K-halves per lane for column N = lane&15, matching the A fragment order)
// -------------------------------------------------------------------------
__global__ void swizzle_weights(const float* __restrict__ W,
                                const float* __restrict__ U,
                                _Float16* __restrict__ wq) {
  int fid = blockIdx.x * blockDim.x + threadIdx.x;   // one half per thread
  int kk   = fid & 15;
  int lane = (fid >> 4) & 31;
  int nt   = (fid >> 9) & 127;
  int kt   = fid >> 16;                              // NT*32*16 == 65536
  if (kt >= KT) return;
  int k   = kt * 32 + ((lane & 16) ? 8 : 0) + (kk & 7) + ((kk & 8) ? 16 : 0);
  int col = nt * 16 + (lane & 15);
  float v = (k < F_) ? W[k * N_ + col] : U[(k - F_) * N_ + col];
  wq[fid] = (_Float16)v;
}

// -------------------------------------------------------------------------
// Persistent LSTM: one workgroup per 16-row batch tile, 32 waves.
// Wave w owns hidden columns [16w,16w+16): gate tiles nt = {w, w+32, w+64, w+96}.
// c (fp32) and masked h (fp32) live in accumulator-layout VGPRs across all T
// steps; x_{t+1} is prefetched into registers during step t's GEMM so the
// serial T-loop never stalls on a fresh global load.
// -------------------------------------------------------------------------
__global__ __launch_bounds__(1024, 1)
void lstm_persistent(const float* __restrict__ x,
                     const _Float16* __restrict__ wq,
                     const float* __restrict__ b,
                     const float* __restrict__ Wd,
                     const float* __restrict__ bd,
                     float* __restrict__ out) {
  __shared__ __align__(32) _Float16 Abuf[16 * LDA];  // A = [x_t | h], fp16
  __shared__ __align__(8)  unsigned int maskhalf[32];
  __shared__ __align__(16) float hbuf[16 * H_];      // final h, fp32

  const int tid  = threadIdx.x;
  const int lane = tid & 31;
  const int w    = tid >> 5;           // wave id 0..31
  const int bg   = blockIdx.x;         // batch tile 0..15
  const int ln15 = lane & 15;
  const int hi8  = (lane & 16) ? 8 : 0;

  // initial h = 0 in the A buffer's h columns
  for (int i = tid; i < 16 * H_; i += 1024) {
    int r = i >> 9, c = i & (H_ - 1);
    Abuf[r * LDA + F_ + c] = (_Float16)0.0f;
  }

  // per-wave gate biases: column = g*H_ + w*16 + (lane&15), same for all 8 rows
  const float bz0 = b[0 * H_ + w * 16 + ln15];
  const float bz1 = b[1 * H_ + w * 16 + ln15];
  const float bz2 = b[2 * H_ + w * 16 + ln15];
  const float bz3 = b[3 * H_ + w * 16 + ln15];

  // B-fragment base pointers (per kt stride = NT*32*16 = 65536 halves)
  const _Float16* wq0 = wq + (((0 * 32 + w) * 32 + lane) << 4);
  const _Float16* wq1 = wq + (((1 * 32 + w) * 32 + lane) << 4);
  const _Float16* wq2 = wq + (((2 * 32 + w) * 32 + lane) << 4);
  const _Float16* wq3 = wq + (((3 * 32 + w) * 32 + lane) << 4);

  // x loading: 1024 threads * 2 floats = one 16x128 tile per step
  const int xrow = tid >> 6;                    // 0..15
  const int xf   = (tid << 1) & 127;            // 0,2,..,126
  const f2v* xp =
      (const f2v*)(x + ((size_t)(bg * 16 + xrow) * T_) * F_ + xf);

  // A-fragment base: row = lane&15, K base = (lane<16?0:8)  (16-bit A layout)
  const _Float16* Ar = Abuf + ln15 * LDA + hi8;

  float cst[8], hst[8];
#pragma unroll
  for (int r = 0; r < 8; ++r) { cst[r] = 0.0f; hst[r] = 0.0f; }

  // prime the x pipeline (non-temporal: x is streamed exactly once; keep
  // the step-resident weights hot in L2 instead)
  f2v xv = __builtin_nontemporal_load(&xp[0]);

  for (int t = 0; t < T_; ++t) {
    // ---- phase 1: x_t (prefetched last step) -> LDS fp16, half-row mask ----
    Abuf[xrow * LDA + xf]     = (_Float16)xv[0];
    Abuf[xrow * LDA + xf + 1] = (_Float16)xv[1];
    int nz = (__ballot((xv[0] != -1.0f) || (xv[1] != -1.0f)) != 0ull) ? 1 : 0;
    if (lane == 0) maskhalf[w] = (unsigned)nz;
    __syncthreads();  // B1: x, mask, and prev-step h all visible

    // issue the x_{t+1} load now; it overlaps the whole GEMM + gate phase
    {
      int tn = (t + 1 < T_) ? (t + 1) : t;
      xv = __builtin_nontemporal_load(&xp[tn * 64]);
    }

    // mask bits for my 8 rows (read early; maskhalf rewritten next step)
    unsigned m8 = 0;
#pragma unroll
    for (int r = 0; r < 8; ++r) {
      uint2 mm = *(const uint2*)&maskhalf[2 * (r + hi8)];
      if (mm.x | mm.y) m8 |= (1u << r);
    }

    // ---- phase 2: z = [x_t | h] @ [W;U]  (16x640 @ 640x2048, my 4 tiles) ----
    v8f a0 = {}, a1 = {}, a2 = {}, a3 = {};
#pragma unroll 4
    for (int kt = 0; kt < KT; ++kt) {
      V16 a;
      a.h[0] = *(const v8h*)(Ar + kt * 32);
      a.h[1] = *(const v8h*)(Ar + kt * 32 + 16);
      v16h bf0 = *(const v16h*)(wq0 + kt * 65536);
      v16h bf1 = *(const v16h*)(wq1 + kt * 65536);
      v16h bf2 = *(const v16h*)(wq2 + kt * 65536);
      v16h bf3 = *(const v16h*)(wq3 + kt * 65536);
      a0 = __builtin_amdgcn_wmma_f32_16x16x32_f16(false, a.v, false, bf0,
                                                  (short)0, a0, false, false);
      a1 = __builtin_amdgcn_wmma_f32_16x16x32_f16(false, a.v, false, bf1,
                                                  (short)0, a1, false, false);
      a2 = __builtin_amdgcn_wmma_f32_16x16x32_f16(false, a.v, false, bf2,
                                                  (short)0, a2, false, false);
      a3 = __builtin_amdgcn_wmma_f32_16x16x32_f16(false, a.v, false, bf3,
                                                  (short)0, a3, false, false);
    }
    __syncthreads();  // B2: all A reads done; safe to overwrite h (and x next iter)

    // ---- phase 3: gates + cell update in registers, write h back fp16 ----
#pragma unroll
    for (int r = 0; r < 8; ++r) {
      float zi = a0[r] + bz0;
      float zf = a1[r] + bz1;
      float zg = a2[r] + bz2;
      float zo = a3[r] + bz3;
      float ig = sigmoidf_(zi);
      float fg = sigmoidf_(zf);
      float gg = fmaxf(zg, 0.0f);
      float og = sigmoidf_(zo);
      float cn = fg * cst[r] + ig * gg;
      float hn = og * fmaxf(cn, 0.0f);
      if ((m8 >> r) & 1u) { cst[r] = cn; hst[r] = hn; }
      Abuf[(r + hi8) * LDA + F_ + w * 16 + ln15] = (_Float16)hst[r];
    }
  }

  // ---- final dense head: out = sigmoid(h @ Wd + bd), one wave per row ----
#pragma unroll
  for (int r = 0; r < 8; ++r)
    hbuf[(r + hi8) * H_ + w * 16 + ln15] = hst[r];
  __syncthreads();

  if (w < 16) {
    float s = 0.0f;
#pragma unroll
    for (int kk = 0; kk < 16; ++kk) {
      int c = lane + (kk << 5);
      s += hbuf[w * H_ + c] * Wd[c];
    }
#pragma unroll
    for (int off = 16; off > 0; off >>= 1)
      s += __shfl_down(s, off, 32);
    if (lane == 0) out[bg * 16 + w] = sigmoidf_(s + bd[0]);
  }
}

// -------------------------------------------------------------------------
extern "C" void kernel_launch(void* const* d_in, const int* in_sizes, int n_in,
                              void* d_out, int out_size, void* d_ws,
                              size_t ws_size, hipStream_t stream) {
  const float* x  = (const float*)d_in[0];   // (B,T,F)
  const float* W  = (const float*)d_in[1];   // (F,4H)
  const float* U  = (const float*)d_in[2];   // (H,4H)
  const float* b  = (const float*)d_in[3];   // (4H,)
  const float* Wd = (const float*)d_in[4];   // (H,1)
  const float* bd = (const float*)d_in[5];   // (1,)
  _Float16* wq = (_Float16*)d_ws;            // 640*2048*2B = 2.62 MB swizzled weights

  // pack [W;U] into per-lane WMMA B fragments (fp16), L2-resident thereafter
  swizzle_weights<<<(KT * NT * 32 * 16) / 256, 256, 0, stream>>>(W, U, wq);

  // 16 persistent workgroups (one per 16-row batch tile), 32 waves each
  lstm_persistent<<<B_ / 16, 1024, 0, stream>>>(x, wq, b, Wd, bd, (float*)d_out);
}
